// FlowLSTM_5609227288937
// MI455X (gfx1250) — compile-verified, matrix operand order checked by reading, not run on hardware
//
#include <hip/hip_runtime.h>

typedef _Float16 h16;
typedef __attribute__((ext_vector_type(16))) _Float16 v16h;
typedef __attribute__((ext_vector_type(8)))  _Float16 v8h;
typedef __attribute__((ext_vector_type(8)))  float    v8f;
typedef __attribute__((ext_vector_type(4)))  unsigned int u4;

#define T_LEN 19
#define I_DIM 17

// ---------------- workspace layout (half-element offsets) ----------------
enum : int {
  HO_WIH0 = 0,                      // 512 x 40  (K=17 padded to 32, stride 40)
  HO_WHH0 = HO_WIH0 + 512 * 40,     // 512 x 136 (K=128, stride 136)
  HO_WIH1 = HO_WHH0 + 512 * 136,
  HO_WHH1 = HO_WIH1 + 512 * 136,
  HO_WIH2 = HO_WHH1 + 512 * 136,
  HO_WHH2 = HO_WIH2 + 512 * 136,
  HO_WL   = HO_WHH2 + 512 * 136,    // 32 x 136  (N=17 padded to 32)
  HO_END  = HO_WL + 32 * 136,
};
#define BPACK_OFF (HO_END * 2)      // byte offset of f32 bias pack (3*512 + 32 floats)

// ---------------- LDS layout (byte offsets) ----------------
enum : int {
  L_WIH = 0,                         // 512*136*2 = 139264 (layer0 uses stride 40 inside)
  L_WHH = L_WIH + 512 * 136 * 2,     // 139264
  L_WL  = L_WHH + 512 * 136 * 2,     // 8704
  L_X   = L_WL  + 32 * 136 * 2,      // 64 rows x 40 halves = 5120
  L_SCR = L_X   + 64 * 40 * 2,       // 4 waves x 16 x 24 halves = 3072
  L_B   = L_SCR + 4 * 16 * 24 * 2,   // 3*512 f32 = 6144
  L_BL  = L_B   + 3 * 512 * 4,       // 32 f32 = 128
  L_END = L_BL  + 32 * 4,            // total 301696 bytes  (< 320KB WGP LDS)
};

// ---------------- fast activations (hardware transcendentals) ----------------
__device__ __forceinline__ float fsig(float x) {
  return __builtin_amdgcn_rcpf(1.0f + __builtin_amdgcn_exp2f(-1.442695040888963f * x));
}
#if __has_builtin(__builtin_amdgcn_tanhf)
__device__ __forceinline__ float ftanh(float x) { return __builtin_amdgcn_tanhf(x); }
#else
__device__ __forceinline__ float ftanh(float x) {
  float e = __builtin_amdgcn_exp2f(2.885390081777927f * x);   // e^{2x}
  return (e - 1.0f) * __builtin_amdgcn_rcpf(e + 1.0f);
}
#endif

// Load a 32x16 (KxN) B-fragment -- or a 16x32 A-fragment -- from LDS row-major
// storage. Lane L: row = base_row + (L&15), K-group = (L>>4)*8; halves 0..7 are
// K=kg..kg+7, halves 8..15 are K=16+kg..16+kg+7 (per CDNA5 WMMA VGPR layout).
__device__ __forceinline__ v16h load_frag(const h16* base, int row, int stride_h,
                                          int kbase, int lane) {
  const h16* p = base + row * stride_h + kbase + ((lane >> 4) << 3);
  v8h lo = *(const v8h*)p;
  v8h hi = *(const v8h*)(p + 16);
  v16h r;
#pragma unroll
  for (int i = 0; i < 8; ++i) { r[i] = lo[i]; r[i + 8] = hi[i]; }
  return r;
}

__device__ __forceinline__ v8f wmma16(v16h a, v16h b, v8f c) {
  return __builtin_amdgcn_wmma_f32_16x16x32_f16(false, a, false, b, (short)0, c,
                                                false, false);
}

// ---------------- prep: f32 -> padded f16 weight pack + fused biases ----------------
__global__ void lstm_prep(const float* Wih0, const float* Whh0, const float* bih0, const float* bhh0,
                          const float* Wih1, const float* Whh1, const float* bih1, const float* bhh1,
                          const float* Wih2, const float* Whh2, const float* bih2, const float* bhh2,
                          const float* Wl, const float* bl,
                          h16* wp, float* bp) {
  int idx = blockIdx.x * blockDim.x + threadIdx.x;
  int st  = gridDim.x * blockDim.x;
  for (int i = idx; i < 512 * 40; i += st) {
    int n = i / 40, k = i - 40 * n;
    wp[HO_WIH0 + i] = (h16)((k < 17) ? Wih0[n * 17 + k] : 0.0f);
  }
  for (int i = idx; i < 512 * 136; i += st) {
    int n = i / 136, k = i - 136 * n;
    float v0 = (k < 128) ? Whh0[n * 128 + k] : 0.0f;
    float v1 = (k < 128) ? Wih1[n * 128 + k] : 0.0f;
    float v2 = (k < 128) ? Whh1[n * 128 + k] : 0.0f;
    float v3 = (k < 128) ? Wih2[n * 128 + k] : 0.0f;
    float v4 = (k < 128) ? Whh2[n * 128 + k] : 0.0f;
    wp[HO_WHH0 + i] = (h16)v0;
    wp[HO_WIH1 + i] = (h16)v1;
    wp[HO_WHH1 + i] = (h16)v2;
    wp[HO_WIH2 + i] = (h16)v3;
    wp[HO_WHH2 + i] = (h16)v4;
  }
  for (int i = idx; i < 32 * 136; i += st) {
    int n = i / 136, k = i - 136 * n;
    wp[HO_WL + i] = (h16)(((n < 17) && (k < 128)) ? Wl[n * 128 + k] : 0.0f);
  }
  for (int i = idx; i < 512; i += st) {
    bp[i]        = bih0[i] + bhh0[i];
    bp[512 + i]  = bih1[i] + bhh1[i];
    bp[1024 + i] = bih2[i] + bhh2[i];
  }
  for (int i = idx; i < 32; i += st) bp[1536 + i] = (i < 17) ? bl[i] : 0.0f;
}

// ---------------- fused 3-layer LSTM + projection ----------------
// grid: B/64 blocks; block: 128 threads = 4 wave32; each wave owns 16 batch rows.
__global__ __launch_bounds__(128)
void lstm3_fused(const float* __restrict__ x, const h16* __restrict__ wp,
                 const float* __restrict__ bp, float* __restrict__ out) {
  extern __shared__ char smem[];
  h16*   Wih = (h16*)(smem + L_WIH);
  h16*   Whh = (h16*)(smem + L_WHH);
  h16*   Wls = (h16*)(smem + L_WL);
  h16*   xb  = (h16*)(smem + L_X);
  h16*   scr = (h16*)(smem + L_SCR);
  float* bia = (float*)(smem + L_B);
  float* blp = (float*)(smem + L_BL);

  const int tid  = threadIdx.x;
  const int wave = tid >> 5;
  const int lane = tid & 31;
  const int ln15 = lane & 15;
  const int lhi  = lane >> 4;
  const int rowbase = blockIdx.x * 64 + wave * 16;

  // one-time stage: Wl tile + fused biases
  {
    const u4* s = (const u4*)(wp + HO_WL);
    u4* d = (u4*)Wls;
    for (int i = tid; i < (32 * 136 * 2) / 16; i += 128) d[i] = s[i];
    const u4* s2 = (const u4*)bp;
    u4* d2 = (u4*)bia;  // bia..blp contiguous, 6272 bytes
    for (int i = tid; i < 6272 / 16; i += 128) d2[i] = s2[i];
  }

  // persistent per-wave state: c in accumulator layout, h in A-fragment layout
  v8f  zero8 = {};
  v16h zeroh = {};
  v8f  c_st[3][8];
  v16h hA[3][4];
#pragma unroll
  for (int l = 0; l < 3; ++l) {
#pragma unroll
    for (int j = 0; j < 8; ++j) c_st[l][j] = zero8;
#pragma unroll
    for (int k = 0; k < 4; ++k) hA[l][k] = zeroh;
  }

  const int wih_off[3]  = {HO_WIH0, HO_WIH1, HO_WIH2};
  const int whh_off[3]  = {HO_WHH0, HO_WHH1, HO_WHH2};
  const int wih_b16[3]  = {512 * 40 * 2 / 16, 512 * 136 * 2 / 16, 512 * 136 * 2 / 16};
  h16* sc = scr + wave * (16 * 24);

  for (int t = 0; t < T_LEN; ++t) {
    __syncthreads();  // previous readers of xb/weights are done
    // stage x[t] tile: 64 rows x 40 halves, cols 17..39 zero
    for (int i = tid; i < 64 * 40; i += 128) {
      int r = i / 40, cc = i - 40 * r;
      float v = (cc < 17)
                  ? x[(size_t)(blockIdx.x * 64 + r) * (T_LEN * I_DIM) + t * I_DIM + cc]
                  : 0.0f;
      xb[i] = (h16)v;
    }

#pragma unroll
    for (int layer = 0; layer < 3; ++layer) {
      __syncthreads();  // all waves done with previous layer's staged weights
      {
        const u4* s = (const u4*)(wp + wih_off[layer]);
        u4* d = (u4*)Wih;
        for (int i = tid; i < wih_b16[layer]; i += 128) d[i] = s[i];
        const u4* s2 = (const u4*)(wp + whh_off[layer]);
        u4* d2 = (u4*)Whh;
        for (int i = tid; i < (512 * 136 * 2) / 16; i += 128) d2[i] = s2[i];
      }
      __syncthreads();  // staged weights visible

      const int lp = (layer > 0) ? layer - 1 : 0;
      v16h hAn[4] = {};

#pragma unroll
      for (int j = 0; j < 8; ++j) {
        const int nj = j * 16;
        // per-gate bias for this chunk's 16 output columns (added post-GEMM;
        // WMMA accumulator chains start from an inline-zero C operand)
        float bv0 = bia[layer * 512 + 0 * 128 + nj + ln15];
        float bv1 = bia[layer * 512 + 1 * 128 + nj + ln15];
        float bv2 = bia[layer * 512 + 2 * 128 + nj + ln15];
        float bv3 = bia[layer * 512 + 3 * 128 + nj + ln15];

        v8f acc[4];
        // input contribution: x (layer 0, K=32) or previous layer's h (K=128)
        if (layer == 0) {
          v16h a = load_frag(xb, wave * 16 + ln15, 40, 0, lane);
#pragma unroll
          for (int g = 0; g < 4; ++g) {
            v16h b = load_frag(Wih, g * 128 + nj + ln15, 40, 0, lane);
            acc[g] = wmma16(a, b, zero8);           // C = inline 0
          }
        } else {
#pragma unroll
          for (int kt = 0; kt < 4; ++kt) {
            v16h a = hA[lp][kt];  // h_{l-1}(t), already updated this timestep
#pragma unroll
            for (int g = 0; g < 4; ++g) {
              v16h b = load_frag(Wih, g * 128 + nj + ln15, 136, kt * 32, lane);
              acc[g] = (kt == 0) ? wmma16(a, b, zero8) : wmma16(a, b, acc[g]);
            }
          }
        }
        // recurrent contribution: h_l(t-1)
#pragma unroll
        for (int kt = 0; kt < 4; ++kt) {
          v16h a = hA[layer][kt];
#pragma unroll
          for (int g = 0; g < 4; ++g) {
            v16h b = load_frag(Whh, g * 128 + nj + ln15, 136, kt * 32, lane);
            acc[g] = wmma16(a, b, acc[g]);
          }
        }
        // elementwise LSTM cell (gate order i,f,g,o); write h chunk to scratch
#pragma unroll
        for (int e = 0; e < 8; ++e) {
          float iv = fsig(acc[0][e] + bv0);
          float fv = fsig(acc[1][e] + bv1);
          float gv = ftanh(acc[2][e] + bv2);
          float ov = fsig(acc[3][e] + bv3);
          float cn = fv * c_st[layer][j][e] + iv * gv;
          c_st[layer][j][e] = cn;
          float hv = ov * ftanh(cn);
          int M = e + (lhi << 3);              // D-layout row
          sc[M * 24 + ln15] = (h16)hv;
        }
        // D-layout -> A-layout via per-wave LDS scratch. LDS ops from one wave
        // execute in order (ISA ch.5), so the RAW is safe without a DS drain;
        // the empty asm only pins the compiler's ordering.
        asm volatile("" ::: "memory");
        v8h hh = *(const v8h*)(sc + ln15 * 24 + (lhi << 3));
        const int kt = j >> 1;
        if (j & 1) {
#pragma unroll
          for (int e = 0; e < 8; ++e) hAn[kt][8 + e] = hh[e];
        } else {
#pragma unroll
          for (int e = 0; e < 8; ++e) hAn[kt][e] = hh[e];
        }
      }  // chunks
#pragma unroll
      for (int kt = 0; kt < 4; ++kt) hA[layer][kt] = hAn[kt];
    }  // layers

    // output projection: out[:,t,:] = h2 @ Wl^T + bl  (N=17 padded to 32)
#pragma unroll
    for (int nt = 0; nt < 2; ++nt) {
      float bv = blp[nt * 16 + ln15];
      v8f acc;
#pragma unroll
      for (int kt = 0; kt < 4; ++kt) {
        v16h b = load_frag(Wls, nt * 16 + ln15, 136, kt * 32, lane);
        acc = (kt == 0) ? wmma16(hA[2][kt], b, zero8) : wmma16(hA[2][kt], b, acc);
      }
      int ci = nt * 16 + ln15;
      if (ci < I_DIM) {
#pragma unroll
        for (int e = 0; e < 8; ++e) {
          int M = e + (lhi << 3);
          out[(size_t)(rowbase + M) * (T_LEN * I_DIM) + t * I_DIM + ci] = acc[e] + bv;
        }
      }
    }
  }  // t
}

extern "C" void kernel_launch(void* const* d_in, const int* in_sizes, int n_in,
                              void* d_out, int out_size, void* d_ws, size_t ws_size,
                              hipStream_t stream) {
  (void)in_sizes; (void)n_in; (void)out_size; (void)ws_size;
  const float* x    = (const float*)d_in[0];
  const float* Wih0 = (const float*)d_in[1];
  const float* Whh0 = (const float*)d_in[2];
  const float* bih0 = (const float*)d_in[3];
  const float* bhh0 = (const float*)d_in[4];
  const float* Wih1 = (const float*)d_in[5];
  const float* Whh1 = (const float*)d_in[6];
  const float* bih1 = (const float*)d_in[7];
  const float* bhh1 = (const float*)d_in[8];
  const float* Wih2 = (const float*)d_in[9];
  const float* Whh2 = (const float*)d_in[10];
  const float* bih2 = (const float*)d_in[11];
  const float* bhh2 = (const float*)d_in[12];
  const float* Wl   = (const float*)d_in[13];
  const float* bl   = (const float*)d_in[14];

  h16*   wp = (h16*)d_ws;
  float* bp = (float*)((char*)d_ws + BPACK_OFF);

  lstm_prep<<<256, 256, 0, stream>>>(Wih0, Whh0, bih0, bhh0,
                                     Wih1, Whh1, bih1, bhh1,
                                     Wih2, Whh2, bih2, bhh2,
                                     Wl, bl, wp, bp);

  // 16384 / 64 rows per WG = 256 workgroups, 128 threads (4 wave32) each
  lstm3_fused<<<256, 128, L_END, stream>>>(x, wp, bp, (float*)d_out);
}
